// MyGlobalAttention_3307124818005
// MI455X (gfx1250) — matrix operand order repaired and checked by reading, not compiled, not run on hardware
//
#include <hip/hip_runtime.h>

// Problem dims (fixed by the reference)
#define D_BATCH 4096
#define NUM_ANN 8192
#define D_QUERY 1024
#define D_ANN   1024

// GEMM tiling
#define BLK_M 128
#define BLK_N 128
#define KSTEP 64
#define LDS_STRIDE 72   // 64 data elems + 8 pad -> 144B rows (16B aligned, conflict-free)
#define LDS_TILE   (BLK_M * LDS_STRIDE)            // elems per (matrix, buffer)
#define GEMM_LDS_BYTES (4 * LDS_TILE * 2)          // A/B x double buffer, bf16

typedef __attribute__((ext_vector_type(8)))  __bf16 bf16x8;
typedef __attribute__((ext_vector_type(16))) __bf16 bf16x16;
typedef __attribute__((ext_vector_type(8)))  float  f32x8;
typedef __attribute__((ext_vector_type(4)))  float  f32x4;
typedef __attribute__((ext_vector_type(4)))  unsigned short u16x4;
typedef __attribute__((ext_vector_type(4)))  unsigned int   u32x4;
typedef __attribute__((ext_vector_type(8)))  int            i32x8;
typedef __attribute__((ext_vector_type(4)))  int            i32x4;

// ---------- helpers ----------
__device__ __forceinline__ unsigned short f2bf(float f) {
    unsigned u = __float_as_uint(f);
    u += 0x7FFFu + ((u >> 16) & 1u);           // round-to-nearest-even
    return (unsigned short)(u >> 16);
}

// ---- Tensor Data Mover: DMA a [BLK_M x KSTEP] bf16 tile (row stride K elems)
// from global memory into LDS with 16B padding after each 128B row
// (-> LDS row stride = 144B = LDS_STRIDE elems). ISA 08_async_tensor §8.3/8.4.
__device__ __forceinline__ void tdm_load_tile(unsigned lds_off, const void* gaddr, int K) {
    const unsigned long long ga = (unsigned long long)gaddr;
    u32x4 g0;
    g0[0] = 1u;                                              // count=1, user load descriptor
    g0[1] = lds_off;                                         // lds_addr (bytes)
    g0[2] = (unsigned)ga;                                    // global_addr[31:0]
    g0[3] = (unsigned)((ga >> 32) & 0x01FFFFFFu) | (2u << 30); // addr[56:32] | type=2 (image)
    i32x8 g1;
    g1[0] = (1 << 16)      // data_size = 2 bytes
          | (1 << 20)      // pad_enable
          | (4 << 22)      // pad_interval: 2^(4+1)=32 DWORDs = 128B between pads
          | (3 << 25);     // pad_amount: 4 DWORDs = 16B
    g1[1] = (KSTEP & 0xFFFF) << 16;   // tensor_dim0 = 64 (bits 79:48, low half)
    g1[2] = (BLK_M & 0xFFFF) << 16;   // tensor_dim0 hi=0 | tensor_dim1 = 128 (low half)
    g1[3] = (KSTEP & 0xFFFF) << 16;   // tensor_dim1 hi=0 | tile_dim0 = 64
    g1[4] = BLK_M;                    // tile_dim1 = 128 | tile_dim2 = 0
    g1[5] = K;                        // tensor_dim0_stride[31:0] (elems)
    g1[6] = 0;                        // stride[47:32]=0 | tensor_dim1_stride lo = 0
    g1[7] = 0;
    const i32x4 z4 = {0, 0, 0, 0};
    const i32x8 z8 = {0, 0, 0, 0, 0, 0, 0, 0};
    __builtin_amdgcn_tensor_load_to_lds(g0, g1, z4, z4, z8, 0);
}

// ---------- f32 -> bf16 bulk convert (vectorized x4) ----------
__global__ __launch_bounds__(256)
void k_cvt_bf16(const float* __restrict__ src, unsigned short* __restrict__ dst, int n4) {
    int i = blockIdx.x * 256 + threadIdx.x;
    if (i >= n4) return;
    f32x4 v = ((const f32x4*)src)[i];
    u16x4 o;
    o[0] = f2bf(v[0]); o[1] = f2bf(v[1]); o[2] = f2bf(v[2]); o[3] = f2bf(v[3]);
    ((u16x4*)dst)[i] = o;
}

// ---------- f32 [rows x cols] -> bf16 transpose [cols x rows], LDS tiled ----------
__global__ __launch_bounds__(256)
void k_transpose_bf16(const float* __restrict__ src, unsigned short* __restrict__ dst,
                      int rows, int cols) {
    __shared__ unsigned short tile[32][34];
    const int c0 = blockIdx.x * 32, r0 = blockIdx.y * 32;
    const int tx = threadIdx.x, ty = threadIdx.y;   // block = (32, 8)
#pragma unroll
    for (int i = 0; i < 32; i += 8)
        tile[ty + i][tx] = f2bf(src[(size_t)(r0 + ty + i) * cols + (c0 + tx)]);
    __syncthreads();
#pragma unroll
    for (int i = 0; i < 32; i += 8)
        dst[(size_t)(c0 + ty + i) * rows + (r0 + tx)] = tile[tx][ty + i];
}

// ---------- block-tiled NT GEMM: C[M,N] = A[M,K](bf16) @ B[N,K](bf16)^T ----------
// Block = 256 threads = 8 waves computes a 128x128 tile.
// Per KSTEP=64 stage: TDM (wave 0) DMAs A(128x64) and B(128x64) bf16 tiles into
// padded LDS (double-buffered; s_wait_tensorcnt + s_barrier handoff); every wave
// computes a 32x64 sub-tile = 16 x v_wmma_f32_16x16x32_bf16 fed by ds_load_b128.
template <int WITH_BIAS, int OUT_BF16>
__global__ __launch_bounds__(256)
void k_gemm_nt(const unsigned short* __restrict__ A,
               const unsigned short* __restrict__ B,
               const float* __restrict__ bias,
               float* __restrict__ Cf,
               unsigned short* __restrict__ Cb,
               int M, int N, int K) {
    extern __shared__ unsigned short smem[];   // 73,728B: [A0][A1][B0][B1]

    const int tid  = threadIdx.x;
    const int lane = tid & 31;
    const int wave = tid >> 5;
    const int l15  = lane & 15;
    const int hi   = lane >> 4;
    const int wm   = wave >> 1;     // 0..3 -> 32-row strips
    const int wn   = wave & 1;      // 0..1 -> 64-col strips

    const int nBlocksN = N / BLK_N;
    const int bm = blockIdx.x / nBlocksN;
    const int bn = blockIdx.x - bm * nBlocksN;
    const int rowA0 = bm * BLK_M;
    const int rowB0 = bn * BLK_N;

    const unsigned short* gA = A + (size_t)rowA0 * K;
    const unsigned short* gB = B + (size_t)rowB0 * K;

    auto offA = [&](int b) { return b * LDS_TILE; };
    auto offB = [&](int b) { return 2 * LDS_TILE + b * LDS_TILE; };

    auto stage = [&](int buf, int k0) {   // wave-0-only TDM issue (EXEC-ignored DMA)
        tdm_load_tile((unsigned)(unsigned long long)(smem + offA(buf)), gA + k0, K);
        tdm_load_tile((unsigned)(unsigned long long)(smem + offB(buf)), gB + k0, K);
    };

    f32x8 acc[2][4] = {};

    const int nStages = K / KSTEP;
    if (wave == 0) stage(0, 0);
    for (int s = 0; s < nStages; ++s) {
        if (wave == 0) __builtin_amdgcn_s_wait_tensorcnt(0);   // DMA into buf (s&1) done
        __syncthreads();                                       // visible to all waves;
                                                               // prior reads of alt done
        if (wave == 0 && s + 1 < nStages) stage((s + 1) & 1, (s + 1) * KSTEP);

        const unsigned short* bufA = smem + offA(s & 1);
        const unsigned short* bufB = smem + offB(s & 1);

        // Fragments per ISA 7.12.2 (wave32, 16-bit):
        //  A 16x32: lanes 0-15 hold K=0..7 & 16..23; lanes 16-31 K=8..15 & 24..31
        //  B 32x16: lanes 0-15 hold K=0..7 & 16..23; lanes 16-31 K=8..15 & 24..31
#pragma unroll
        for (int kc = 0; kc < KSTEP; kc += 32) {
            bf16x16 a[2], b[4];
#pragma unroll
            for (int mt = 0; mt < 2; ++mt) {
                const unsigned short* p =
                    bufA + (wm * 32 + mt * 16 + l15) * LDS_STRIDE + kc + hi * 8;
                bf16x8 lo = *(const bf16x8*)p;
                bf16x8 hh = *(const bf16x8*)(p + 16);
#pragma unroll
                for (int i = 0; i < 8; ++i) { a[mt][i] = lo[i]; a[mt][i + 8] = hh[i]; }
            }
#pragma unroll
            for (int nt = 0; nt < 4; ++nt) {
                const unsigned short* p =
                    bufB + (wn * 64 + nt * 16 + l15) * LDS_STRIDE + kc + hi * 8;
                bf16x8 lo = *(const bf16x8*)p;
                bf16x8 hh = *(const bf16x8*)(p + 16);
#pragma unroll
                for (int i = 0; i < 8; ++i) { b[nt][i] = lo[i]; b[nt][i + 8] = hh[i]; }
            }
#pragma unroll
            for (int mt = 0; mt < 2; ++mt)
#pragma unroll
                for (int nt = 0; nt < 4; ++nt)
                    acc[mt][nt] = __builtin_amdgcn_wmma_f32_16x16x32_bf16(
                        false, a[mt], false, b[nt], (short)0, acc[mt][nt], false, false);
        }
    }

    // Epilogue: D layout — VGPR r holds (m = r + 8*hi, n = lane&15)
#pragma unroll
    for (int mt = 0; mt < 2; ++mt) {
        const int rbase = rowA0 + wm * 32 + mt * 16 + hi * 8;
#pragma unroll
        for (int nt = 0; nt < 4; ++nt) {
            const int col = rowB0 + wn * 64 + nt * 16 + l15;
            const float badd = WITH_BIAS ? bias[col] : 0.0f;
#pragma unroll
            for (int r = 0; r < 8; ++r) {
                const float v = acc[mt][nt][r] + badd;
                const size_t off = (size_t)(rbase + r) * N + col;
                if (OUT_BF16) Cb[off] = f2bf(v);
                else          Cf[off] = v;
            }
        }
    }
}

// ---------- row softmax (in place, f32) + bf16 copy for the context GEMM ----------
__global__ __launch_bounds__(256)
void k_softmax(float* __restrict__ attn, unsigned short* __restrict__ attn16, int ncols) {
    __shared__ float red[256];
    const int row = blockIdx.x;
    float* rp = attn + (size_t)row * ncols;

    float m = -3.402823466e38f;
    for (int c = threadIdx.x; c < ncols; c += 256) m = fmaxf(m, rp[c]);
    red[threadIdx.x] = m; __syncthreads();
    for (int s = 128; s > 0; s >>= 1) {
        if (threadIdx.x < s) red[threadIdx.x] = fmaxf(red[threadIdx.x], red[threadIdx.x + s]);
        __syncthreads();
    }
    m = red[0]; __syncthreads();

    float sum = 0.f;
    for (int c = threadIdx.x; c < ncols; c += 256) {
        float e = __expf(rp[c] - m);
        rp[c] = e;
        sum += e;
    }
    red[threadIdx.x] = sum; __syncthreads();
    for (int s = 128; s > 0; s >>= 1) {
        if (threadIdx.x < s) red[threadIdx.x] += red[threadIdx.x + s];
        __syncthreads();
    }
    const float inv = 1.0f / red[0];

    unsigned short* rp16 = attn16 + (size_t)row * ncols;
    for (int c = threadIdx.x; c < ncols; c += 256) {
        float w = rp[c] * inv;
        rp[c]   = w;
        rp16[c] = f2bf(w);
    }
}

// ---------- launch ----------
extern "C" void kernel_launch(void* const* d_in, const int* in_sizes, int n_in,
                              void* d_out, int out_size, void* d_ws, size_t ws_size,
                              hipStream_t stream) {
    (void)in_sizes; (void)n_in; (void)out_size; (void)ws_size;

    const float* query = (const float*)d_in[0];   // [4096,1024]
    const float* ann   = (const float*)d_in[1];   // [8192,1024]
    const float* fc_w  = (const float*)d_in[2];   // [1024,1024]
    const float* fc_b  = (const float*)d_in[3];   // [1024]

    float* context = (float*)d_out;                                   // [4096,1024]
    float* attn    = (float*)d_out + (size_t)D_BATCH * D_ANN;         // [4096,8192]

    // Workspace layout (bf16 elements)
    unsigned short* ws16 = (unsigned short*)d_ws;
    const size_t O_Q16 = 0;                                   // q bf16        [4096x1024]
    const size_t O_QRY = O_Q16 + (size_t)D_BATCH * D_QUERY;   // query bf16    [4096x1024]
    const size_t O_FCW = O_QRY + (size_t)D_BATCH * D_QUERY;   // fc_w bf16     [1024x1024]
    const size_t O_ANN = O_FCW + (size_t)D_ANN * D_QUERY;     // ann bf16      [8192x1024]
    const size_t O_ANT = O_ANN + (size_t)NUM_ANN * D_ANN;     // ann^T bf16    [1024x8192]
    const size_t O_ATT = O_ANT + (size_t)NUM_ANN * D_ANN;     // attn bf16     [4096x8192]

    // 1) convert inputs to bf16 (+ transposed annotations for the context GEMM)
    k_cvt_bf16<<<D_BATCH * D_QUERY / 4 / 256, 256, 0, stream>>>(query, ws16 + O_QRY, D_BATCH * D_QUERY / 4);
    k_cvt_bf16<<<D_ANN * D_QUERY / 4 / 256, 256, 0, stream>>>(fc_w, ws16 + O_FCW, D_ANN * D_QUERY / 4);
    k_cvt_bf16<<<NUM_ANN * D_ANN / 4 / 256, 256, 0, stream>>>(ann, ws16 + O_ANN, NUM_ANN * D_ANN / 4);
    k_transpose_bf16<<<dim3(D_ANN / 32, NUM_ANN / 32), dim3(32, 8), 0, stream>>>(ann, ws16 + O_ANT, NUM_ANN, D_ANN);

    // 2) q = query @ fc_w^T + fc_b -> bf16        (M=4096, N=1024, K=1024)
    k_gemm_nt<1, 1><<<(D_BATCH / BLK_M) * (D_ANN / BLK_N), 256, GEMM_LDS_BYTES, stream>>>(
        ws16 + O_QRY, ws16 + O_FCW, fc_b, nullptr, ws16 + O_Q16, D_BATCH, D_ANN, D_QUERY);

    // 3) logits = q @ ann^T -> f32 (attn region)  (M=4096, N=8192, K=1024)
    k_gemm_nt<0, 0><<<(D_BATCH / BLK_M) * (NUM_ANN / BLK_N), 256, GEMM_LDS_BYTES, stream>>>(
        ws16 + O_Q16, ws16 + O_ANN, nullptr, attn, nullptr, D_BATCH, NUM_ANN, D_ANN);

    // 4) softmax rows in place + bf16 copy
    k_softmax<<<D_BATCH, 256, 0, stream>>>(attn, ws16 + O_ATT, NUM_ANN);

    // 5) context = attn_bf16 @ annT^T -> f32      (M=4096, N=1024, K=8192)
    k_gemm_nt<0, 0><<<(D_BATCH / BLK_M) * (D_ANN / BLK_N), 256, GEMM_LDS_BYTES, stream>>>(
        ws16 + O_ATT, ws16 + O_ANT, nullptr, context, nullptr, D_BATCH, D_ANN, NUM_ANN);
}
